// EdgeGATLayer_31525059953122
// MI455X (gfx1250) — compile-verified, matrix operand order checked by reading, not run on hardware
//
#include <hip/hip_runtime.h>

typedef __attribute__((ext_vector_type(16))) __bf16 v16bf;
typedef __attribute__((ext_vector_type(8)))  float  v8f;

#define KN 32          // neighbors per node
#define DD 128         // feature dim (in == out)
#define STRIDE 132     // padded LDS row stride (floats): bank-spread + 16B row alignment
#define GRP 8          // nodes per wave-group (h_t batched through one WMMA tile)
#define NEG_SLOPE 0.2f
#define RESID 0.12f

union Frag { unsigned u[8]; v16bf v; };

// split f32 pair into packed bf16 hi and bf16 residual-lo words (3xBF16 f32 emulation)
__device__ __forceinline__ void splitPack(float x0, float x1, unsigned &hi, unsigned &lo) {
  unsigned b0 = __float_as_uint(x0), b1 = __float_as_uint(x1);
  unsigned h0 = b0 & 0xFFFF0000u,    h1 = b1 & 0xFFFF0000u;
  float r0 = x0 - __uint_as_float(h0);
  float r1 = x1 - __uint_as_float(h1);
  hi = (h0 >> 16) | h1;
  lo = (__float_as_uint(r0) >> 16) | (__float_as_uint(r1) & 0xFFFF0000u);
}

__device__ __forceinline__ float leaky(float x) { return x > 0.f ? x : NEG_SLOPE * x; }

__global__ __launch_bounds__(128) void gat_fused_kernel(
    const float* __restrict__ srcH,   // [N, 32, 128]
    const float* __restrict__ hIn,    // [N, 128]
    const float* __restrict__ W,      // [128, 128] (in,out)
    const float* __restrict__ aSrc,   // [128]
    const float* __restrict__ aDst,   // [128]
    float* __restrict__ out,          // [N, 128]
    int nNodes)
{
  // W pre-swizzled into exact B-fragment lane layout, split-bf16 (hi/lo)
  __shared__ __align__(16) unsigned sWh[4][8][32][8];   // [kt][nt][lane][vgpr] 32KB
  __shared__ __align__(16) unsigned sWl[4][8][32][8];   // 32KB
  __shared__ __align__(16) float sAs[DD];
  __shared__ __align__(16) float sAd[DD];
  __shared__ __align__(16) float sBuf[4][KN * STRIDE];  // per-wave node tile (reused for z)
  __shared__ __align__(16) float sHt[4][GRP * DD];      // per-wave cached h_t rows

  const int tid  = threadIdx.x;
  const int wave = tid >> 5;
  const int lane = tid & 31;
  const int half = lane >> 4;
  const int l15  = lane & 15;

  // ---- one-time per block: pack W (hi/lo bf16) into B-fragment layout ----
  for (int idx = tid; idx < 4 * 8 * 32; idx += blockDim.x) {
    int lkt   = idx >> 8;
    int lnt   = (idx >> 5) & 7;
    int lln   = idx & 31;
    int lhalf = lln >> 4;
    int n     = lnt * 16 + (lln & 15);
    #pragma unroll
    for (int v = 0; v < 8; ++v) {
      int base = (v < 4 ? 2 * v : 16 + 2 * (v - 4)) + 8 * lhalf;
      int k0   = lkt * 32 + base;
      float x0 = W[k0 * DD + n];
      float x1 = W[(k0 + 1) * DD + n];
      unsigned hi, lo;
      splitPack(x0, x1, hi, lo);
      sWh[lkt][lnt][lln][v] = hi;
      sWl[lkt][lnt][lln][v] = lo;
    }
  }
  if (tid < DD) { sAs[tid] = aSrc[tid]; sAd[tid] = aDst[tid]; }
  __syncthreads();

  float* buf  = sBuf[wave];
  float* hbuf = sHt[wave];
  const int wavesTotal = gridDim.x * 4;
  const int nGroups    = (nNodes + GRP - 1) / GRP;

  for (int grp = blockIdx.x * 4 + wave; grp < nGroups; grp += wavesTotal) {
    const int base   = grp * GRP;
    const int nvalid = min(GRP, nNodes - base);

    // ======== phase 1: h_t = leaky(h @ W) for the whole group (one 16-row tile) ========
    #pragma unroll
    for (int r = 0; r < GRP; ++r) {
      float4 g = {0.f, 0.f, 0.f, 0.f};
      if (r < nvalid) g = *(const float4*)&hIn[(size_t)(base + r) * DD + lane * 4];
      *(float4*)&buf[r * STRIDE + lane * 4] = g;
    }
    {
      v8f hacc[8];
      #pragma unroll
      for (int nt = 0; nt < 8; ++nt)
        #pragma unroll
        for (int i = 0; i < 8; ++i) hacc[nt][i] = 0.f;

      for (int kt = 0; kt < 4; ++kt) {
        Frag ah, al;
        const bool valid = (l15 < GRP);
        const int  m     = l15 & (GRP - 1);
        #pragma unroll
        for (int v = 0; v < 8; ++v) {
          int vb = (v < 4 ? 2 * v : 16 + 2 * (v - 4)) + 8 * half;
          int k0 = kt * 32 + vb;
          float x0 = 0.f, x1 = 0.f;
          if (valid) {
            const float2 p = *(const float2*)&buf[m * STRIDE + k0];
            x0 = p.x; x1 = p.y;
          }
          splitPack(x0, x1, ah.u[v], al.u[v]);
        }
        #pragma unroll
        for (int nt = 0; nt < 8; ++nt) {
          Frag bh, bl;
          *(uint4*)&bh.u[0] = *(const uint4*)&sWh[kt][nt][lane][0];
          *(uint4*)&bh.u[4] = *(const uint4*)&sWh[kt][nt][lane][4];
          *(uint4*)&bl.u[0] = *(const uint4*)&sWl[kt][nt][lane][0];
          *(uint4*)&bl.u[4] = *(const uint4*)&sWl[kt][nt][lane][4];
          hacc[nt] = __builtin_amdgcn_wmma_f32_16x16x32_bf16(
              false, ah.v, false, bh.v, (short)0, hacc[nt], false, false);
          hacc[nt] = __builtin_amdgcn_wmma_f32_16x16x32_bf16(
              false, ah.v, false, bl.v, (short)0, hacc[nt], false, false);
          hacc[nt] = __builtin_amdgcn_wmma_f32_16x16x32_bf16(
              false, al.v, false, bh.v, (short)0, hacc[nt], false, false);
        }
      }
      // leaky + store valid h_t rows (rows 0..7 live in half==0 lanes)
      #pragma unroll
      for (int nt = 0; nt < 8; ++nt)
        #pragma unroll
        for (int r = 0; r < 8; ++r) {
          float x = leaky(hacc[nt][r]);
          if (half == 0) hbuf[r * DD + nt * 16 + l15] = x;
        }
    }

    // ======== phase 2: per node in group ========
    for (int g = 0; g < nvalid; ++g) {
      const int node = base + g;

      // stream src_h[node] (32x128 f32) into LDS, coalesced float4 per row
      #pragma unroll 4
      for (int r = 0; r < KN; ++r) {
        const float4 gg = *(const float4*)&srcH[((size_t)node * KN + r) * DD + lane * 4];
        *(float4*)&buf[r * STRIDE + lane * 4] = gg;
      }
      if (node + 1 < nNodes)
        __builtin_prefetch(&srcH[(size_t)(node + 1) * KN * DD + lane * 32], 0, 1);

      // GEMM: z[32x128] = leaky(tile @ W), 3xBF16 split
      v8f acc[2][8];
      #pragma unroll
      for (int mt = 0; mt < 2; ++mt)
        #pragma unroll
        for (int nt = 0; nt < 8; ++nt)
          #pragma unroll
          for (int i = 0; i < 8; ++i) acc[mt][nt][i] = 0.f;

      for (int kt = 0; kt < 4; ++kt) {
        Frag ah[2], al[2];
        #pragma unroll
        for (int mt = 0; mt < 2; ++mt) {
          const int m = mt * 16 + l15;
          #pragma unroll
          for (int v = 0; v < 8; ++v) {
            int vb = (v < 4 ? 2 * v : 16 + 2 * (v - 4)) + 8 * half;
            int k0 = kt * 32 + vb;
            const float2 p = *(const float2*)&buf[m * STRIDE + k0];
            splitPack(p.x, p.y, ah[mt].u[v], al[mt].u[v]);
          }
        }
        #pragma unroll
        for (int nt = 0; nt < 8; ++nt) {
          Frag bh, bl;
          *(uint4*)&bh.u[0] = *(const uint4*)&sWh[kt][nt][lane][0];
          *(uint4*)&bh.u[4] = *(const uint4*)&sWh[kt][nt][lane][4];
          *(uint4*)&bl.u[0] = *(const uint4*)&sWl[kt][nt][lane][0];
          *(uint4*)&bl.u[4] = *(const uint4*)&sWl[kt][nt][lane][4];
          #pragma unroll
          for (int mt = 0; mt < 2; ++mt) {
            acc[mt][nt] = __builtin_amdgcn_wmma_f32_16x16x32_bf16(
                false, ah[mt].v, false, bh.v, (short)0, acc[mt][nt], false, false);
            acc[mt][nt] = __builtin_amdgcn_wmma_f32_16x16x32_bf16(
                false, ah[mt].v, false, bl.v, (short)0, acc[mt][nt], false, false);
            acc[mt][nt] = __builtin_amdgcn_wmma_f32_16x16x32_bf16(
                false, al[mt].v, false, bh.v, (short)0, acc[mt][nt], false, false);
          }
        }
      }

      // LeakyReLU + write z in-place over the LDS tile (src fully consumed)
      #pragma unroll
      for (int mt = 0; mt < 2; ++mt)
        #pragma unroll
        for (int nt = 0; nt < 8; ++nt)
          #pragma unroll
          for (int r = 0; r < 8; ++r) {
            const int m = mt * 16 + r + 8 * half;
            buf[m * STRIDE + nt * 16 + l15] = leaky(acc[mt][nt][r]);
          }

      // c = h_t . a_dst (wave reduction); h_t row from group cache
      const float4 hrow = *(const float4*)&hbuf[g * DD + lane * 4];
      const float4 ad4  = *(const float4*)&sAd[lane * 4];
      float cpart = hrow.x * ad4.x + hrow.y * ad4.y + hrow.z * ad4.z + hrow.w * ad4.w;
      #pragma unroll
      for (int off = 16; off > 0; off >>= 1) cpart += __shfl_xor(cpart, off, 32);

      // e[k] = leaky(z[k] . a_src + c); one neighbor per lane
      float e = 0.f;
      #pragma unroll
      for (int j = 0; j < 32; ++j) {
        const float4 z4 = *(const float4*)&buf[lane * STRIDE + j * 4];
        const float4 a4 = *(const float4*)&sAs[j * 4];
        e += z4.x * a4.x + z4.y * a4.y + z4.z * a4.z + z4.w * a4.w;
      }
      e = leaky(e + cpart);

      // softmax over the 32 lanes
      float mx = e;
      #pragma unroll
      for (int off = 16; off > 0; off >>= 1) mx = fmaxf(mx, __shfl_xor(mx, off, 32));
      const float p = __expf(e - mx);
      float s = p;
      #pragma unroll
      for (int off = 16; off > 0; off >>= 1) s += __shfl_xor(s, off, 32);
      const float alpha = p / s;

      // msg = sum_k alpha_k * z[k]; 4 columns per lane; add residual
      float ax = 0.f, ay = 0.f, az = 0.f, aw = 0.f;
      #pragma unroll
      for (int k = 0; k < 32; ++k) {
        const float ak  = __shfl(alpha, k, 32);
        const float4 z4 = *(const float4*)&buf[k * STRIDE + lane * 4];
        ax += ak * z4.x; ay += ak * z4.y; az += ak * z4.z; aw += ak * z4.w;
      }
      float4 res;
      res.x = ax + RESID * hrow.x;
      res.y = ay + RESID * hrow.y;
      res.z = az + RESID * hrow.z;
      res.w = aw + RESID * hrow.w;
      *(float4*)&out[(size_t)node * DD + lane * 4] = res;
    }
  }
}

extern "C" void kernel_launch(void* const* d_in, const int* in_sizes, int n_in,
                              void* d_out, int out_size, void* d_ws, size_t ws_size,
                              hipStream_t stream) {
  const float* srcH = (const float*)d_in[0];
  const float* hIn  = (const float*)d_in[1];
  const float* W    = (const float*)d_in[2];
  const float* aS   = (const float*)d_in[3];
  const float* aD   = (const float*)d_in[4];
  float* out = (float*)d_out;
  int nNodes = out_size / DD;

  int nGroups = (nNodes + GRP - 1) / GRP;
  int blocks  = (nGroups + 3) / 4;
  if (blocks > 768) blocks = 768;
  if (blocks < 1) blocks = 1;
  gat_fused_kernel<<<blocks, 128, 0, stream>>>(srcH, hIn, W, aS, aD, out, nNodes);
}
